// ConvResidualBlock_28767690948628
// MI455X (gfx1250) — matrix operand order repaired and checked by reading, not compiled, not run on hardware
//
#include <hip/hip_runtime.h>

typedef __attribute__((ext_vector_type(16))) _Float16 v16h;
typedef __attribute__((ext_vector_type(8)))  float    v8f;

#define CIN  128
#define COUT 64

// ---------------------------------------------------------------- utilities
__global__ __launch_bounds__(256) void zero_f32(float* __restrict__ p, long long n) {
  long long i = (long long)blockIdx.x * blockDim.x + threadIdx.x;
  long long stride = (long long)gridDim.x * blockDim.x;
  for (; i < n; i += stride) p[i] = 0.0f;
}

// deg[dst] += 1 per edge (edge_index is [2,E] int32: row0=src, row1=dst)
__global__ __launch_bounds__(256) void degree_kernel(const int* __restrict__ ei,
                                                     float* __restrict__ deg, int E) {
  int i = blockIdx.x * blockDim.x + threadIdx.x;
  int stride = gridDim.x * blockDim.x;
  for (; i < E; i += stride) atomicAdd(&deg[ei[E + i]], 1.0f);
}

// deg -> deg^{-1/2} with self-loop (+1)
__global__ __launch_bounds__(256) void dinv_kernel(float* __restrict__ deg, int N) {
  int i = blockIdx.x * blockDim.x + threadIdx.x;
  if (i < N) deg[i] = rsqrtf(deg[i] + 1.0f);
}

// One-shot fp32 -> f16 conversion of both weight matrices (done once per
// launch instead of once per GEMM block: removes ~400k redundant v_cvt ops).
__global__ __launch_bounds__(256) void cvt_weights_kernel(
    const float* __restrict__ W, const float* __restrict__ Wr,
    _Float16* __restrict__ wh, _Float16* __restrict__ wrh, int n /* 64*128 */) {
  int i = blockIdx.x * blockDim.x + threadIdx.x;
  if (i < n) {
    wh[i]  = (_Float16)W[i];
    wrh[i] = (_Float16)Wr[i];
  }
}

// ------------------------------------------------- fused dual GEMM via WMMA
// h = x @ W^T            -> ws
// r = x @ Wr^T + br      -> d_out (residual, consumed by final kernel)
// 256 threads = 8 waves; wave w owns rows [blk*128 + 16w, +16), all 64 cols.
__global__ __launch_bounds__(256) void gemm_wmma_kernel(
    const float*    __restrict__ x,    // [N,128] fp32
    const _Float16* __restrict__ wh,   // [64,128] f16 (pre-converted)
    const _Float16* __restrict__ wrh,  // [64,128] f16 (pre-converted)
    const float*    __restrict__ br,   // [64]
    float* __restrict__ h,             // [N,64]
    float* __restrict__ r,             // [N,64]
    int N)
{
  const int lane = threadIdx.x & 31;
  const int wave = threadIdx.x >> 5;
  const int row0 = blockIdx.x * 128 + wave * 16;
  const int m    = lane & 15;   // M (A) / N (B,D) index within tile
  const int half = lane >> 4;   // lane-group selector for K striping

  // Clamp (branchless) so EXEC stays all-ones for WMMA; stores are guarded.
  int rowA = row0 + m;
  rowA = rowA < N ? rowA : (N - 1);
  const float* __restrict__ xrow = x + (size_t)rowA * CIN;

  v8f zero = {};
  v8f acc_h[4] = {zero, zero, zero, zero};
  v8f acc_r[4] = {zero, zero, zero, zero};

#pragma unroll
  for (int kk = 0; kk < 4; ++kk) {            // K = 128 in 4 steps of 32
    const int k0 = kk * 32;
    // A fragment: 16x32 f16, contiguous K order matching B
    // (lanes 0-15: K=0..15 ; lanes 16-31: K=16..31), element j -> K=half*16+j.
    v16h a;
    {
      const float* xp = xrow + k0 + half * 16;
#pragma unroll
      for (int j = 0; j < 8; ++j) a[j]     = (_Float16)xp[j];
#pragma unroll
      for (int j = 0; j < 8; ++j) a[8 + j] = (_Float16)xp[8 + j];
    }
#pragma unroll
    for (int ct = 0; ct < 4; ++ct) {          // 4 col tiles of 16 -> 64 cols
      // B fragment: 32x16 f16, lane = N, element j -> K = half*16 + j.
      // B[k][n] = W[ct*16+n][k0+k]; 16 consecutive f16 = 32B aligned load.
      const size_t boff = (size_t)(ct * 16 + m) * CIN + k0 + half * 16;
      const v16h bw  = *(const v16h*)(wh  + boff);
      const v16h bwr = *(const v16h*)(wrh + boff);
      acc_h[ct] = __builtin_amdgcn_wmma_f32_16x16x32_f16(
          false, a, false, bw,  (short)0, acc_h[ct], false, false);
      acc_r[ct] = __builtin_amdgcn_wmma_f32_16x16x32_f16(
          false, a, false, bwr, (short)0, acc_r[ct], false, false);
    }
  }

  // D layout: VGPR v -> M = v + 8*half, N = lane&15.
#pragma unroll
  for (int ct = 0; ct < 4; ++ct) {
    const int col = ct * 16 + m;
    const float brv = br[col];
#pragma unroll
    for (int v = 0; v < 8; ++v) {
      const int row = row0 + v + half * 8;
      if (row < N) {
        h[(size_t)row * COUT + col] = acc_h[ct][v];
        r[(size_t)row * COUT + col] = acc_r[ct][v] + brv;
      }
    }
  }
}

// ------------------------------------------------- edge gather/scatter (L2)
// One wave32 per edge; lane handles 2 channels. h (25.6MB) is L2-resident.
__global__ __launch_bounds__(256) void agg_kernel(
    const int* __restrict__ ei, const float* __restrict__ hbuf,
    const float* __restrict__ dinv, float* __restrict__ agg, int E)
{
  const int lane = threadIdx.x & 31;
  int w = blockIdx.x * (blockDim.x >> 5) + (threadIdx.x >> 5);
  const int nW = gridDim.x * (blockDim.x >> 5);
  for (int e = w; e < E; e += nW) {
    const int s = ei[e];
    const int d = ei[E + e];
    const float coef = dinv[s] * dinv[d];
    const float2 v = ((const float2*)(hbuf + (size_t)s * COUT))[lane];
    float* ap = agg + (size_t)d * COUT + 2 * lane;
    atomicAdd(ap,     v.x * coef);
    atomicAdd(ap + 1, v.y * coef);
  }
}

// ---------------------------- self-loop + conv bias + BN statistics (fused)
__global__ __launch_bounds__(256) void selfloop_stats_kernel(
    const float* __restrict__ h, const float* __restrict__ dinv,
    const float* __restrict__ b, float* __restrict__ agg,
    float* __restrict__ stats /* sum[64] | sumsq[64] */, int N)
{
  const int c = threadIdx.x & 63;
  const int g = threadIdx.x >> 6;                 // 4 row-groups per block
  const float bc = b[c];
  float s = 0.0f, s2 = 0.0f;
  for (int i = blockIdx.x * 4 + g; i < N; i += gridDim.x * 4) {
    const float di = dinv[i];
    const size_t idx = (size_t)i * COUT + c;
    const float v = agg[idx] + h[idx] * (di * di) + bc;
    agg[idx] = v;
    s += v; s2 += v * v;
  }
  __shared__ float ls[256], ls2[256];
  ls[threadIdx.x] = s; ls2[threadIdx.x] = s2;
  __syncthreads();
  if (g == 0) {
    s  = ls[c]  + ls[64 + c]  + ls[128 + c]  + ls[192 + c];
    s2 = ls2[c] + ls2[64 + c] + ls2[128 + c] + ls2[192 + c];
    atomicAdd(&stats[c], s);
    atomicAdd(&stats[64 + c], s2);
  }
}

// ------------------------------------ BN normalize + ReLU + residual add
__global__ __launch_bounds__(256) void final_kernel(
    const float* __restrict__ agg, const float* __restrict__ stats,
    const float* __restrict__ gamma, const float* __restrict__ beta,
    float* __restrict__ out /* in: residual r, out: result */, long long total, float invN)
{
  long long i = (long long)blockIdx.x * blockDim.x + threadIdx.x;
  const long long stride = (long long)gridDim.x * blockDim.x;
  for (; i < total; i += stride) {
    const int c = (int)(i & 63);
    const float mean = stats[c] * invN;
    const float var  = stats[64 + c] * invN - mean * mean;  // biased var
    const float y = gamma[c] * (agg[i] - mean) * rsqrtf(var + 1e-5f) + beta[c];
    out[i] = fmaxf(y, 0.0f) + out[i];
  }
}

// --------------------------------------------------------------- launcher
extern "C" void kernel_launch(void* const* d_in, const int* in_sizes, int n_in,
                              void* d_out, int out_size, void* d_ws, size_t ws_size,
                              hipStream_t stream) {
  const float* x     = (const float*)d_in[0];
  const int*   ei    = (const int*)  d_in[1];
  const float* W     = (const float*)d_in[2];
  const float* b     = (const float*)d_in[3];
  const float* gamma = (const float*)d_in[4];
  const float* beta  = (const float*)d_in[5];
  const float* Wr    = (const float*)d_in[6];
  const float* br    = (const float*)d_in[7];
  float* out = (float*)d_out;

  const int N = in_sizes[0] / CIN;
  const int E = in_sizes[1] / 2;
  const int WN = COUT * CIN;                   // 8192 weights per matrix

  float* deg   = (float*)d_ws;                 // [N]
  float* h     = deg + N;                      // [N*64]
  float* agg   = h + (size_t)N * COUT;         // [N*64]
  float* stats = agg + (size_t)N * COUT;       // [128]
  _Float16* wh  = (_Float16*)(stats + 128);    // [64*128] f16
  _Float16* wrh = wh + WN;                     // [64*128] f16

  zero_f32<<<1024, 256, 0, stream>>>(deg, (long long)N);
  zero_f32<<<2048, 256, 0, stream>>>(agg, (long long)N * COUT + 128);

  cvt_weights_kernel<<<(WN + 255) / 256, 256, 0, stream>>>(W, Wr, wh, wrh, WN);

  degree_kernel<<<2048, 256, 0, stream>>>(ei, deg, E);
  dinv_kernel<<<(N + 255) / 256, 256, 0, stream>>>(deg, N);

  gemm_wmma_kernel<<<(N + 127) / 128, 256, 0, stream>>>(x, wh, wrh, br, h, out, N);

  agg_kernel<<<4096, 256, 0, stream>>>(ei, h, deg, agg, E);

  selfloop_stats_kernel<<<1024, 256, 0, stream>>>(h, deg, b, agg, stats, N);

  final_kernel<<<2048, 256, 0, stream>>>(agg, stats, gamma, beta, out,
                                         (long long)N * COUT, 1.0f / (float)N);
}
